// AttentionModule_11424613007426
// MI455X (gfx1250) — matrix operand order repaired and checked by reading, not compiled, not run on hardware
//
#include <hip/hip_runtime.h>
#include <cstdint>

typedef __bf16 bf16;
typedef __attribute__((ext_vector_type(16))) __bf16 v16bf;
typedef __attribute__((ext_vector_type(8)))  float  v8f;

union Frag { v16bf v; uint4 q[2]; };

static __device__ __forceinline__ v8f vzero8() {
  v8f r;
#pragma unroll
  for (int i = 0; i < 8; ++i) r[i] = 0.f;
  return r;
}

static __device__ __forceinline__ float gelu_exact(float x) {
  return 0.5f * x * (1.f + erff(x * 0.70710678118654752f));
}

// ---------------------------------------------------------------------------
// CDNA5 async global->LDS copy (ASYNCcnt-tracked, bypasses VGPRs) with a
// synchronous fallback if the builtin is unavailable on this toolchain.
// The builtin's pointer params are int __vector(4) with AS1/AS3 qualifiers.
// ---------------------------------------------------------------------------
typedef int v4i_plain __attribute__((vector_size(16)));
typedef __attribute__((address_space(1))) v4i_plain g_v4i;
typedef __attribute__((address_space(3))) v4i_plain l_v4i;

#if __has_builtin(__builtin_amdgcn_global_load_async_to_lds_b128)
#define HAS_ASYNC_LDS 1
static __device__ __forceinline__ void async_cp16(void* lds, const void* g) {
  __builtin_amdgcn_global_load_async_to_lds_b128((g_v4i*)g, (l_v4i*)lds, 0, 0);
}
#else
#define HAS_ASYNC_LDS 0
static __device__ __forceinline__ void async_cp16(void* lds, const void* g) {
  *(uint4*)lds = *(const uint4*)g;
}
#endif

static __device__ __forceinline__ void wait_async() {
#if HAS_ASYNC_LDS
#if __has_builtin(__builtin_amdgcn_s_wait_asynccnt)
  __builtin_amdgcn_s_wait_asynccnt(0);
#else
  asm volatile("s_wait_asynccnt 0x0" ::: "memory");
#endif
#endif
}

// ---------------------------------------------------------------------------
// Weight convert+transpose: W[K][N] fp32 -> Wt[N][K] bf16
// ---------------------------------------------------------------------------
__global__ __launch_bounds__(256) void wconv_kernel(const float* __restrict__ W,
                                                    bf16* __restrict__ Wt,
                                                    int K, int N) {
  size_t i = (size_t)blockIdx.x * 256 + threadIdx.x;
  if (i < (size_t)K * N) {
    size_t k = i / N, n = i - k * (size_t)N;
    Wt[n * (size_t)K + k] = (bf16)W[i];
  }
}

__global__ __launch_bounds__(256) void zero_bf16_kernel(bf16* __restrict__ p, long n) {
  long i = (long)blockIdx.x * 256 + threadIdx.x;
  if (i < n) p[i] = (bf16)0.f;
}

// ---------------------------------------------------------------------------
// Embedding build + LayerNorm.  grid = Mpad rows, 256 threads (3 elems/thread)
// ---------------------------------------------------------------------------
__global__ __launch_bounds__(256) void embed_ln_kernel(
    const float* __restrict__ x1, const float* __restrict__ x2,
    const float* __restrict__ cls, const float* __restrict__ sep,
    const float* __restrict__ pos, const float* __restrict__ tt,
    const float* __restrict__ gam, const float* __restrict__ bet,
    float* __restrict__ h, bf16* __restrict__ h16, int L1, int S) {
  const int r = blockIdx.x, tid = threadIdx.x;
  const int Mv = 2 * S;
  if (r >= Mv) {  // pad rows -> zeros so GEMM A-tiles are clean
    for (int i = tid; i < 768; i += 256) {
      h[(size_t)r * 768 + i] = 0.f;
      h16[(size_t)r * 768 + i] = (bf16)0.f;
    }
    return;
  }
  const int b = r / S, s = r - b * S;
  const float* src;
  if (s == 0)             src = cls;
  else if (s <= L1)       src = x1 + ((size_t)b * L1 + (s - 1)) * 768;
  else if (s == L1 + 1)   src = sep;
  else if (s <= 2 * L1 + 1) src = x2 + ((size_t)b * L1 + (s - L1 - 2)) * 768;
  else                    src = sep;

  float val[3];
  float lsum = 0.f;
#pragma unroll
  for (int e = 0; e < 3; ++e) {
    int i = tid + e * 256;
    val[e] = src[i] + pos[(size_t)s * 768 + i] + tt[i];
    lsum += val[e];
  }
  __shared__ float red[256];
  red[tid] = lsum; __syncthreads();
  for (int st = 128; st > 0; st >>= 1) { if (tid < st) red[tid] += red[tid + st]; __syncthreads(); }
  float mean = red[0] / 768.f;
  __syncthreads();
  float lv = 0.f;
#pragma unroll
  for (int e = 0; e < 3; ++e) { float d = val[e] - mean; lv += d * d; }
  red[tid] = lv; __syncthreads();
  for (int st = 128; st > 0; st >>= 1) { if (tid < st) red[tid] += red[tid + st]; __syncthreads(); }
  float rstd = rsqrtf(red[0] / 768.f + 1e-5f);
#pragma unroll
  for (int e = 0; e < 3; ++e) {
    int i = tid + e * 256;
    float y = (val[e] - mean) * rstd * gam[i] + bet[i];
    h[(size_t)r * 768 + i] = y;
    h16[(size_t)r * 768 + i] = (bf16)y;
  }
}

// ---------------------------------------------------------------------------
// Residual add + LayerNorm: h = LN(h + t); writes h (fp32) and h16 (bf16)
// ---------------------------------------------------------------------------
__global__ __launch_bounds__(256) void add_ln_kernel(
    float* __restrict__ h, const float* __restrict__ t,
    const float* __restrict__ gam, const float* __restrict__ bet,
    bf16* __restrict__ h16, int S) {
  const int r = blockIdx.x, tid = threadIdx.x;
  if (r >= 2 * S) return;  // keep pad rows zero
  float val[3];
  float lsum = 0.f;
#pragma unroll
  for (int e = 0; e < 3; ++e) {
    int i = tid + e * 256;
    val[e] = h[(size_t)r * 768 + i] + t[(size_t)r * 768 + i];
    lsum += val[e];
  }
  __shared__ float red[256];
  red[tid] = lsum; __syncthreads();
  for (int st = 128; st > 0; st >>= 1) { if (tid < st) red[tid] += red[tid + st]; __syncthreads(); }
  float mean = red[0] / 768.f;
  __syncthreads();
  float lv = 0.f;
#pragma unroll
  for (int e = 0; e < 3; ++e) { float d = val[e] - mean; lv += d * d; }
  red[tid] = lv; __syncthreads();
  for (int st = 128; st > 0; st >>= 1) { if (tid < st) red[tid] += red[tid + st]; __syncthreads(); }
  float rstd = rsqrtf(red[0] / 768.f + 1e-5f);
#pragma unroll
  for (int e = 0; e < 3; ++e) {
    int i = tid + e * 256;
    float y = (val[e] - mean) * rstd * gam[i] + bet[i];
    h[(size_t)r * 768 + i] = y;
    h16[(size_t)r * 768 + i] = (bf16)y;
  }
}

// ---------------------------------------------------------------------------
// bf16 WMMA GEMM: C[Mpad,N] = A[Mpad,K] * W  (+bias, optional GELU)
// Bt is W transposed: [N][K].  Block = 256 thr (8 waves), tile 128x128,
// each wave owns a 32x64 sub-tile (2x4 WMMA accumulators), K step 32.
// Double-buffered LDS fed by async global->LDS copies: tile k+1 streams in
// while WMMAs consume tile k; one s_wait_asynccnt + one barrier per K-step.
// ---------------------------------------------------------------------------
template <int ACT>  // 0 = none, 1 = GELU
__global__ __launch_bounds__(256) void gemm_kernel(
    const bf16* __restrict__ A, const bf16* __restrict__ Bt,
    const float* __restrict__ bias,
    float* __restrict__ Cf, bf16* __restrict__ Cb, int K, int N) {
  const int nBase = blockIdx.x * 128;
  const int mBase = blockIdx.y * 128;
  __shared__ __align__(16) bf16 As[2][128][40];
  __shared__ __align__(16) bf16 Bs[2][128][40];
  const int tid = threadIdx.x, lane = tid & 31, wv = tid >> 5;
  const int mw = (wv & 3) * 32;
  const int nw = (wv >> 2) * 64;
  const int koffA = (lane < 16) ? 0 : 8;
  const int koffB = (lane < 16) ? 0 : 16;
  const int arow0 = mw + (lane & 15);
  const int bcol0 = nw + (lane & 15);
  const int cr = tid >> 1;          // copy row 0..127
  const int cc = (tid & 1) * 16;    // copy col chunk base (elems)

  v8f acc[2][4];
#pragma unroll
  for (int mt = 0; mt < 2; ++mt)
#pragma unroll
    for (int nt = 0; nt < 4; ++nt) acc[mt][nt] = vzero8();

  auto load_tiles = [&](int buf, int k0) {
    const bf16* ga = &A[(size_t)(mBase + cr) * K + k0 + cc];
    async_cp16(&As[buf][cr][cc],     ga);
    async_cp16(&As[buf][cr][cc + 8], ga + 8);
    const bf16* gb = &Bt[(size_t)(nBase + cr) * K + k0 + cc];
    async_cp16(&Bs[buf][cr][cc],     gb);
    async_cp16(&Bs[buf][cr][cc + 8], gb + 8);
  };

  load_tiles(0, 0);
  int p = 0;
  for (int k0 = 0; k0 < K; k0 += 32) {
    wait_async();        // this wave's buffer-p tile landed in LDS
    __syncthreads();     // all waves done loading p / done computing p^1
    if (k0 + 32 < K) load_tiles(p ^ 1, k0 + 32);  // prefetch next tile
#pragma unroll
    for (int mt = 0; mt < 2; ++mt) {
      Frag a;
      a.q[0] = *(const uint4*)&As[p][arow0 + mt * 16][koffA];
      a.q[1] = *(const uint4*)&As[p][arow0 + mt * 16][16 + koffA];
#pragma unroll
      for (int nt = 0; nt < 4; ++nt) {
        Frag bfr;
        bfr.q[0] = *(const uint4*)&Bs[p][bcol0 + nt * 16][koffA];
        bfr.q[1] = *(const uint4*)&Bs[p][bcol0 + nt * 16][16 + koffA];
        acc[mt][nt] = __builtin_amdgcn_wmma_f32_16x16x32_bf16(
            false, a.v, false, bfr.v, (short)0, acc[mt][nt], false, false);
      }
    }
    p ^= 1;
  }
#pragma unroll
  for (int mt = 0; mt < 2; ++mt)
#pragma unroll
    for (int nt = 0; nt < 4; ++nt)
#pragma unroll
      for (int e = 0; e < 8; ++e) {
        int m = mBase + mw + mt * 16 + ((lane < 16) ? e : e + 8);
        int n = nBase + nw + nt * 16 + (lane & 15);
        float x = acc[mt][nt][e] + bias[n];
        if (ACT == 1) x = gelu_exact(x);
        if (Cf) Cf[(size_t)m * N + n] = x;
        if (Cb) Cb[(size_t)m * N + n] = (bf16)x;
      }
}

// ---------------------------------------------------------------------------
// qg projection only for the 6 global-token rows (scaled by 1/8)
// ---------------------------------------------------------------------------
__global__ __launch_bounds__(256) void qg_kernel(
    const bf16* __restrict__ h16, const float* __restrict__ Wqg,
    const float* __restrict__ bqg, float* __restrict__ qg_s, int S, int L1) {
  const int gi = blockIdx.x;                 // 0..5 = b*3+g
  const int b = gi / 3, g = gi - 3 * (gi / 3);
  const int gp = (g == 0) ? 0 : ((g == 1) ? (L1 + 1) : (S - 1));
  const size_t r = (size_t)b * S + gp;
  __shared__ float xs[768];
  for (int i = threadIdx.x; i < 768; i += 256) xs[i] = (float)h16[r * 768 + i];
  __syncthreads();
  for (int n = threadIdx.x; n < 768; n += 256) {
    float acc = bqg[n];
    for (int k = 0; k < 768; ++k) acc += xs[k] * Wqg[(size_t)k * 768 + n];
    qg_s[(size_t)gi * 768 + n] = acc * 0.125f;
  }
}

// ---------------------------------------------------------------------------
// Banded sliding-window attention + global-token columns.
// grid = (C, H, B), 256 threads.  WMMA for QK^T (64x192x64) and P.V (64x64x192)
// ---------------------------------------------------------------------------
__global__ __launch_bounds__(256) void attn_band_kernel(
    const bf16* __restrict__ q16, const bf16* __restrict__ k16,
    const bf16* __restrict__ v16, bf16* __restrict__ a16, int S, int L1) {
  const int cBlk = blockIdx.x, hh = blockIdx.y, b = blockIdx.z;
  const int tid = threadIdx.x, lane = tid & 31, wv = tid >> 5;

  __shared__ __align__(16) char smem[117248];
  bf16 (*Qs)[72]   = reinterpret_cast<bf16(*)[72]>(smem);           //  9216 B
  bf16 (*Ks)[72]   = reinterpret_cast<bf16(*)[72]>(smem + 9216);    // 27648 B
  bf16 (*Pb)[208]  = reinterpret_cast<bf16(*)[208]>(smem + 9216);   // aliases Ks (dead after scores)
  bf16 (*Vts)[200] = reinterpret_cast<bf16(*)[200]>(smem + 36864);  // 25600 B (V transposed)
  float (*Sc)[208] = reinterpret_cast<float(*)[208]>(smem + 62464); // 53248 B
  float (*gv)[64]  = reinterpret_cast<float(*)[64]>(smem + 115712); //   768 B
  float (*gk)[64]  = reinterpret_cast<float(*)[64]>(smem + 116480); //   768 B

  const size_t base = ((size_t)b * S) * 768 + (size_t)hh * 64;
  const uint4 z4 = {0u, 0u, 0u, 0u};

  // ---- stage tiles -------------------------------------------------------
  for (int i = tid; i < 512; i += 256) {   // Q 64x64
    int row = i >> 3, col = (i & 7) * 8;
    int s = cBlk * 64 + row;
    uint4 d = (s < S) ? *(const uint4*)&q16[base + (size_t)s * 768 + col] : z4;
    *(uint4*)&Qs[row][col] = d;
  }
  for (int i = tid; i < 1536; i += 256) {  // K_ext 192x64
    int row = i >> 3, col = (i & 7) * 8;
    int j = cBlk * 64 - 64 + row;
    uint4 d = (j >= 0 && j < S) ? *(const uint4*)&k16[base + (size_t)j * 768 + col] : z4;
    *(uint4*)&Ks[row][col] = d;
  }
  for (int i = tid; i < 12288; i += 256) { // V_ext transposed: Vts[d][kk]
    int kk = i >> 6, dcol = i & 63;
    int j = cBlk * 64 - 64 + kk;
    Vts[dcol][kk] = (j >= 0 && j < S) ? v16[base + (size_t)j * 768 + dcol] : (bf16)0.f;
  }
  for (int i = tid; i < 192; i += 256) {   // global-token k/v rows
    int gi = i >> 6, dcol = i & 63;
    int gp = (gi == 0) ? 0 : ((gi == 1) ? (L1 + 1) : (S - 1));
    gv[gi][dcol] = (float)v16[base + (size_t)gp * 768 + dcol];
    gk[gi][dcol] = (float)k16[base + (size_t)gp * 768 + dcol];
  }
  __syncthreads();

  const int mt = (wv & 3) * 16;
  const int arow = mt + (lane & 15);
  const int koffA = (lane < 16) ? 0 : 8;
  const int koffB = (lane < 16) ? 0 : 16;

  // ---- scores = (Q K_ext^T) * 0.125, masked ------------------------------
  const int nbase = (wv >> 2) * 6;   // waves 0-3: n-tiles 0..5; waves 4-7: 6..11
  v8f sacc[6];
#pragma unroll
  for (int t = 0; t < 6; ++t) sacc[t] = vzero8();
#pragma unroll
  for (int k0 = 0; k0 < 64; k0 += 32) {
    Frag a;
    a.q[0] = *(const uint4*)&Qs[arow][k0 + koffA];
    a.q[1] = *(const uint4*)&Qs[arow][k0 + 16 + koffA];
#pragma unroll
    for (int t = 0; t < 6; ++t) {
      int col = (nbase + t) * 16 + (lane & 15);
      Frag bfr;
      bfr.q[0] = *(const uint4*)&Ks[col][k0 + koffA];
      bfr.q[1] = *(const uint4*)&Ks[col][k0 + 16 + koffA];
      sacc[t] = __builtin_amdgcn_wmma_f32_16x16x32_bf16(
          false, a.v, false, bfr.v, (short)0, sacc[t], false, false);
    }
  }
#pragma unroll
  for (int t = 0; t < 6; ++t)
#pragma unroll
    for (int e = 0; e < 8; ++e) {
      int qi = mt + ((lane < 16) ? e : e + 8);
      int kk = (nbase + t) * 16 + (lane & 15);
      int rel = kk - 64 - qi;
      int j = cBlk * 64 - 64 + kk;
      bool valid = (rel >= -64) && (rel <= 64) && (j >= 0) && (j < S);
      Sc[qi][3 + kk] = valid ? sacc[t][e] * 0.125f : -1.0e9f;
    }
  if (tid < 192) {                         // global-token scores (cols 0..2)
    int qi = tid / 3, gi = tid - 3 * (tid / 3);
    float acc = 0.f;
#pragma unroll
    for (int dcol = 0; dcol < 64; ++dcol) acc += (float)Qs[qi][dcol] * gk[gi][dcol];
    Sc[qi][gi] = acc * 0.125f;
  }
  __syncthreads();

  // ---- softmax over [3 global + 192 band] --------------------------------
  if (tid < 64) {
    int qi = tid;
    float mx = -3.0e38f;
    for (int j2 = 0; j2 < 195; ++j2) mx = fmaxf(mx, Sc[qi][j2]);
    float sum = 0.f;
    for (int j2 = 0; j2 < 195; ++j2) { float e = __expf(Sc[qi][j2] - mx); Sc[qi][j2] = e; sum += e; }
    float inv = 1.f / sum;
    Sc[qi][0] *= inv; Sc[qi][1] *= inv; Sc[qi][2] *= inv;
    for (int kk = 0; kk < 192; ++kk) Pb[qi][kk] = (bf16)(Sc[qi][3 + kk] * inv);
  }
  __syncthreads();

  // ---- out = P_band . V_ext  (+ global part) -----------------------------
  const int ntb = (wv >> 2) * 2;  // waves 0-3: d-tiles 0,1; waves 4-7: 2,3
  v8f pacc[2];
  pacc[0] = vzero8(); pacc[1] = vzero8();
#pragma unroll
  for (int k0 = 0; k0 < 192; k0 += 32) {
    Frag a;
    a.q[0] = *(const uint4*)&Pb[arow][k0 + koffA];
    a.q[1] = *(const uint4*)&Pb[arow][k0 + 16 + koffA];
#pragma unroll
    for (int t = 0; t < 2; ++t) {
      int col = (ntb + t) * 16 + (lane & 15);
      Frag bfr;
      bfr.q[0] = *(const uint4*)&Vts[col][k0 + koffA];
      bfr.q[1] = *(const uint4*)&Vts[col][k0 + 16 + koffA];
      pacc[t] = __builtin_amdgcn_wmma_f32_16x16x32_bf16(
          false, a.v, false, bfr.v, (short)0, pacc[t], false, false);
    }
  }
#pragma unroll
  for (int t = 0; t < 2; ++t)
#pragma unroll
    for (int e = 0; e < 8; ++e) {
      int qi = mt + ((lane < 16) ? e : e + 8);
      int dcol = (ntb + t) * 16 + (lane & 15);
      int s = cBlk * 64 + qi;
      if (s < S) {
        float val = pacc[t][e] + Sc[qi][0] * gv[0][dcol] + Sc[qi][1] * gv[1][dcol] +
                    Sc[qi][2] * gv[2][dcol];
        a16[base + (size_t)s * 768 + dcol] = (bf16)val;
      }
    }
}

// ---------------------------------------------------------------------------
// Global-token full attention (overwrites the gpos rows of a16).
// grid = (G=3, H, B), 256 threads.
// ---------------------------------------------------------------------------
__global__ __launch_bounds__(256) void attn_global_kernel(
    const float* __restrict__ qg_s, const bf16* __restrict__ kg16,
    const bf16* __restrict__ vg16, bf16* __restrict__ a16, int S, int L1) {
  const int gi = blockIdx.x, hh = blockIdx.y, b = blockIdx.z;
  const int gp = (gi == 0) ? 0 : ((gi == 1) ? (L1 + 1) : (S - 1));
  const int tid = threadIdx.x;
  __shared__ float qrow[64];
  __shared__ float red[256];
  __shared__ float accs[256][65];
  const size_t base = ((size_t)b * S) * 768 + (size_t)hh * 64;
  if (tid < 64) qrow[tid] = qg_s[((size_t)(b * 3 + gi)) * 768 + hh * 64 + tid];
  __syncthreads();

  float mloc = -3.0e38f;
  for (int s = tid; s < S; s += 256) {
    const bf16* kr = &kg16[base + (size_t)s * 768];
    float sc = 0.f;
#pragma unroll
    for (int dd = 0; dd < 64; ++dd) sc += qrow[dd] * (float)kr[dd];
    mloc = fmaxf(mloc, sc);
  }
  red[tid] = mloc; __syncthreads();
  for (int st = 128; st > 0; st >>= 1) { if (tid < st) red[tid] = fmaxf(red[tid], red[tid + st]); __syncthreads(); }
  const float M = red[0]; __syncthreads();

  float acc[64];
#pragma unroll
  for (int dd = 0; dd < 64; ++dd) acc[dd] = 0.f;
  float lsum = 0.f;
  for (int s = tid; s < S; s += 256) {
    const bf16* kr = &kg16[base + (size_t)s * 768];
    float sc = 0.f;
#pragma unroll
    for (int dd = 0; dd < 64; ++dd) sc += qrow[dd] * (float)kr[dd];
    float e = __expf(sc - M);
    lsum += e;
    const bf16* vr = &vg16[base + (size_t)s * 768];
#pragma unroll
    for (int dd = 0; dd < 64; ++dd) acc[dd] += e * (float)vr[dd];
  }
  red[tid] = lsum; __syncthreads();
  for (int st = 128; st > 0; st >>= 1) { if (tid < st) red[tid] += red[tid + st]; __syncthreads(); }
  const float tot = red[0]; __syncthreads();
#pragma unroll
  for (int dd = 0; dd < 64; ++dd) accs[tid][dd] = acc[dd];
  __syncthreads();
  if (tid < 64) {
    float o = 0.f;
    for (int i = 0; i < 256; ++i) o += accs[i][tid];
    a16[base + (size_t)gp * 768 + tid] = (bf16)(o / tot);
  }
}

// ---------------------------------------------------------------------------
// Pooler: out[b] = tanh(h[b,0] @ pool_W + pool_b)
// ---------------------------------------------------------------------------
__global__ __launch_bounds__(256) void pool_kernel(
    const float* __restrict__ h, const float* __restrict__ pW,
    const float* __restrict__ pb, float* __restrict__ out, int S) {
  const int b = blockIdx.x;
  __shared__ float xs[768];
  const size_t r = (size_t)b * S;
  for (int i = threadIdx.x; i < 768; i += 256) xs[i] = h[r * 768 + i];
  __syncthreads();
  for (int n = threadIdx.x; n < 768; n += 256) {
    float acc = pb[n];
    for (int k = 0; k < 768; ++k) acc += xs[k] * pW[(size_t)k * 768 + n];
    out[(size_t)b * 768 + n] = tanhf(acc);
  }
}

// ---------------------------------------------------------------------------
extern "C" void kernel_launch(void* const* d_in, const int* in_sizes, int n_in,
                              void* d_out, int out_size, void* d_ws, size_t ws_size,
                              hipStream_t stream) {
  (void)n_in; (void)out_size; (void)ws_size;
  const float* x1    = (const float*)d_in[0];
  const float* x2    = (const float*)d_in[1];
  const float* cls   = (const float*)d_in[2];
  const float* sep   = (const float*)d_in[3];
  const float* pos   = (const float*)d_in[4];
  const float* tt    = (const float*)d_in[5];
  const float* eg    = (const float*)d_in[6];
  const float* eb    = (const float*)d_in[7];
  const float* Wq    = (const float*)d_in[8];
  const float* bq    = (const float*)d_in[9];
  const float* Wk    = (const float*)d_in[10];
  const float* bk    = (const float*)d_in[11];
  const float* Wv    = (const float*)d_in[12];
  const float* bv    = (const float*)d_in[13];
  const float* Wqg   = (const float*)d_in[14];
  const float* bqg   = (const float*)d_in[15];
  const float* Wkg   = (const float*)d_in[16];
  const float* bkg   = (const float*)d_in[17];
  const float* Wvg   = (const float*)d_in[18];
  const float* bvg   = (const float*)d_in[19];
  const float* Wo    = (const float*)d_in[20];
  const float* bo    = (const float*)d_in[21];
  const float* ln1g  = (const float*)d_in[22];
  const float* ln1b  = (const float*)d_in[23];
  const float* Wf1   = (const float*)d_in[24];
  const float* bf1   = (const float*)d_in[25];
  const float* Wf2   = (const float*)d_in[26];
  const float* bf2   = (const float*)d_in[27];
  const float* ln2g  = (const float*)d_in[28];
  const float* ln2b  = (const float*)d_in[29];
  const float* pW    = (const float*)d_in[30];
  const float* pbias = (const float*)d_in[31];

  const int L1   = in_sizes[0] / (2 * 768);   // B = 2 fixed by harness
  const int S    = 2 * L1 + 3;
  const int Mv   = 2 * S;
  const int Mpad = ((Mv + 127) / 128) * 128;
  const int C    = (S + 63) / 64;

  char* ws = (char*)d_ws;
  size_t off = 0;
  auto alloc = [&](size_t bytes) -> void* {
    void* p = ws + off;
    off += (bytes + 255) & ~(size_t)255;
    return p;
  };

  const size_t DD = 589824;    // 768*768
  const size_t DDF = 2359296;  // 768*3072
  bf16* wt_qkv[2][7];          // q,k,v,(qg unused),kg,vg,o  transposed bf16
  bf16* wt_f1[2];
  bf16* wt_f2[2];
  for (int l = 0; l < 2; ++l) {
    for (int j = 0; j < 7; ++j) wt_qkv[l][j] = (bf16*)alloc(DD * 2);
    wt_f1[l] = (bf16*)alloc(DDF * 2);
    wt_f2[l] = (bf16*)alloc(DDF * 2);
  }
  float* h     = (float*)alloc((size_t)Mpad * 768 * 4);
  bf16*  h16   = (bf16*)alloc((size_t)Mpad * 768 * 2);
  bf16*  q16b  = (bf16*)alloc((size_t)Mpad * 768 * 2);
  bf16*  k16b  = (bf16*)alloc((size_t)Mpad * 768 * 2);
  bf16*  v16b  = (bf16*)alloc((size_t)Mpad * 768 * 2);
  bf16*  kg16b = (bf16*)alloc((size_t)Mpad * 768 * 2);
  bf16*  vg16b = (bf16*)alloc((size_t)Mpad * 768 * 2);
  bf16*  a16   = (bf16*)alloc((size_t)Mpad * 768 * 2);
  float* tbuf  = (float*)alloc((size_t)Mpad * 768 * 4);
  bf16*  f1b   = (bf16*)alloc((size_t)Mpad * 3072 * 2);
  float* qg_s  = (float*)alloc(6 * 768 * 4);

  // --- convert / transpose all weights to bf16 [N][K] ---------------------
  const float* Wsrc[7] = {Wq, Wk, Wv, Wqg, Wkg, Wvg, Wo};
  for (int l = 0; l < 2; ++l) {
    for (int j = 0; j < 7; ++j) {
      if (j == 3) continue;  // qg handled by dedicated small kernel (fp32)
      wconv_kernel<<<(int)((DD + 255) / 256), 256, 0, stream>>>(
          Wsrc[j] + (size_t)l * DD, wt_qkv[l][j], 768, 768);
    }
    wconv_kernel<<<(int)((DDF + 255) / 256), 256, 0, stream>>>(
        Wf1 + (size_t)l * DDF, wt_f1[l], 768, 3072);
    wconv_kernel<<<(int)((DDF + 255) / 256), 256, 0, stream>>>(
        Wf2 + (size_t)l * DDF, wt_f2[l], 3072, 768);
  }
  {  // zero a16 pad rows so the out-proj GEMM never reads poison
    long cnt = (long)(Mpad - Mv) * 768;
    if (cnt > 0)
      zero_bf16_kernel<<<(int)((cnt + 255) / 256), 256, 0, stream>>>(
          a16 + (size_t)Mv * 768, cnt);
  }

  embed_ln_kernel<<<Mpad, 256, 0, stream>>>(x1, x2, cls, sep, pos, tt, eg, eb,
                                            h, h16, L1, S);

  dim3 gD(768 / 128, Mpad / 128);
  dim3 gDF(3072 / 128, Mpad / 128);
  for (int l = 0; l < 2; ++l) {
    gemm_kernel<0><<<gD, 256, 0, stream>>>(h16, wt_qkv[l][0], bq + l * 768, nullptr, q16b, 768, 768);
    gemm_kernel<0><<<gD, 256, 0, stream>>>(h16, wt_qkv[l][1], bk + l * 768, nullptr, k16b, 768, 768);
    gemm_kernel<0><<<gD, 256, 0, stream>>>(h16, wt_qkv[l][2], bv + l * 768, nullptr, v16b, 768, 768);
    gemm_kernel<0><<<gD, 256, 0, stream>>>(h16, wt_qkv[l][4], bkg + l * 768, nullptr, kg16b, 768, 768);
    gemm_kernel<0><<<gD, 256, 0, stream>>>(h16, wt_qkv[l][5], bvg + l * 768, nullptr, vg16b, 768, 768);
    qg_kernel<<<6, 256, 0, stream>>>(h16, Wqg + (size_t)l * DD, bqg + l * 768, qg_s, S, L1);

    attn_band_kernel<<<dim3(C, 12, 2), 256, 0, stream>>>(q16b, k16b, v16b, a16, S, L1);
    attn_global_kernel<<<dim3(3, 12, 2), 256, 0, stream>>>(qg_s, kg16b, vg16b, a16, S, L1);

    gemm_kernel<0><<<gD, 256, 0, stream>>>(a16, wt_qkv[l][6], bo + l * 768, tbuf, nullptr, 768, 768);
    add_ln_kernel<<<Mpad, 256, 0, stream>>>(h, tbuf, ln1g + l * 768, ln1b + l * 768, h16, S);

    gemm_kernel<1><<<gDF, 256, 0, stream>>>(h16, wt_f1[l], bf1 + l * 3072, nullptr, f1b, 768, 3072);
    gemm_kernel<0><<<gD, 256, 0, stream>>>(f1b, wt_f2[l], bf2 + l * 768, tbuf, nullptr, 3072, 768);
    add_ln_kernel<<<Mpad, 256, 0, stream>>>(h, tbuf, ln2g + l * 768, ln2b + l * 768, h16, S);
  }

  pool_kernel<<<2, 256, 0, stream>>>(h, pW, pbias, (float*)d_out, S);
}
